// WeatherVAE_13030930776589
// MI455X (gfx1250) — matrix operand order repaired and checked by reading, not compiled, not run on hardware
//
#include <hip/hip_runtime.h>

// ---------------------------------------------------------------------------
// WeatherVAE forward for gfx1250 (MI455X).
// Bandwidth-bound (~100 GFLOP vs ~2 GB fp32 traffic @ 23.3 TB/s): all matmul
// work runs on v_wmma_f32_16x16x32_f16. Wave32: each wave owns a
// 16(Cout) x 16(pixel) tile; 8 waves / 256-thread block.
// K reordered spatial-outer/channel-inner so a 32-chunk has uniform (kh,kw):
// one bounds check + constant-stride channel gather per chunk. Bias staged in
// LDS; epilogue is a single divergence with clustered stores.
// ---------------------------------------------------------------------------

typedef __attribute__((ext_vector_type(16))) _Float16 v16h;
typedef __attribute__((ext_vector_type(8)))  _Float16 v8h;
typedef __attribute__((ext_vector_type(8)))  float    v8f;

__device__ __forceinline__ float silu_f(float v) { return v * (1.0f / (1.0f + __expf(-v))); }

template<int ACT>
__device__ __forceinline__ float act_f(float v) {
  if (ACT == 1) return silu_f(v);
  if (ACT == 2) return tanhf(v);
  return v;
}

// ================= fast conv: requires Cin % 32 == 0 ========================
// K-order: K = (kh*KSZ+kw)*Cin + ci. ACT: 0 none, 1 silu, 2 tanh.
template<int ACT, int KSZ>
__global__ __launch_bounds__(256)
void convf_k(const float* __restrict__ x, const float* __restrict__ w,
             const float* __restrict__ bias, float* __restrict__ y,
             int B, int Cin, int Hin, int Win,
             int Cout, int Hout, int Wout, int stride, int pad)
{
  __shared__ __align__(16) _Float16 sA[2][16 * 32];
  __shared__ float sBias[16];
  const int lane = threadIdx.x & 31;
  const int wave = threadIdx.x >> 5;
  const int half = lane >> 4;
  const int l16  = lane & 15;
  const int coutBase = (int)blockIdx.y * 16;
  const int HWo = Hout * Wout;
  const int HWi = Hin * Win;
  const int Npix = B * HWo;
  const int pix  = ((int)blockIdx.x * 8 + wave) * 16 + l16;
  const bool pixOK = pix < Npix;
  const int pp = pixOK ? pix : 0;
  const int b  = pp / HWo;
  const int r0 = pp - b * HWo;
  const int oh = r0 / Wout, ow = r0 - (r0 / Wout) * Wout;
  const int ohs = oh * stride - pad;
  const int ows = ow * stride - pad;
  const float* __restrict__ xb = x + b * Cin * HWi;
  const int K2 = KSZ * KSZ;
  const int KT = Cin * K2;
  const int nChunks = KT >> 5;           // Cin%32==0 -> KT%32==0

  if (threadIdx.x < 16) {
    int co = coutBase + (int)threadIdx.x;
    sBias[threadIdx.x] = co < Cout ? bias[co] : 0.0f;
  }

  v8f acc = {};
  int r = 0, cib = 0;                    // spatial tap r, channel base
  for (int c = 0; c < nChunks; ++c) {
    const int p = c & 1;
    // ---- stage 16x32 f16 weight tile (branchless, clamped) ----
    {
      const int m0 = (int)threadIdx.x >> 5, kq = (int)threadIdx.x & 31;
#pragma unroll
      for (int it = 0; it < 2; ++it) {
        int m = m0 + it * 8;
        int co = coutBase + m;
        int cc = co < Cout ? co : 0;
        float v = w[(cc * Cin + cib + kq) * K2 + r];     // OIHW flat
        sA[p][m * 32 + kq] = (_Float16)(co < Cout ? v : 0.0f);
      }
    }
    // prefetch next chunk's weight line (global_prefetch_b8)
    if (lane == 0 && c + 1 < nChunks) {
      int rn = r, cn = cib + 32;
      if (cn >= Cin) { cn = 0; ++rn; }
      __builtin_prefetch(&w[(coutBase * Cin + cn) * K2 + rn], 0, 1);
    }
    __syncthreads();

    // ---- A fragment: two contiguous 16B LDS runs per lane ----
    const _Float16* rowp = &sA[p][l16 * 32 + 16 * half];
    v8h a0 = *(const v8h*)(rowp);
    v8h a1 = *(const v8h*)(rowp + 8);
    v16h a;
#pragma unroll
    for (int i = 0; i < 8; ++i) { a[i] = a0[i]; a[i + 8] = a1[i]; }

    // ---- B fragment: one tap (kh,kw) per chunk, channel-stride gather ----
    const int kh = (KSZ == 1) ? 0 : (r / KSZ);
    const int kw = (KSZ == 1) ? 0 : (r - kh * KSZ);
    const int ih = ohs + kh, iw = ows + kw;
    const bool inb = pixOK && ih >= 0 && ih < Hin && iw >= 0 && iw < Win;
    const int ihc = inb ? ih : 0, iwc = inb ? iw : 0;
    const float* __restrict__ bp = xb + (cib + 16 * half) * HWi + ihc * Win + iwc;
    float tmp[16];
#pragma unroll
    for (int j = 0; j < 16; ++j) tmp[j] = bp[j * HWi];   // unconditional, clustered
    v16h bf;
#pragma unroll
    for (int j = 0; j < 16; ++j) bf[j] = (_Float16)(inb ? tmp[j] : 0.0f);

    acc = __builtin_amdgcn_wmma_f32_16x16x32_f16(false, a, false, bf,
                                                 (short)0, acc, false, false);
    cib += 32; if (cib >= Cin) { cib = 0; ++r; }
  }

  // D layout: VGPR rr -> M = rr + 8*half, N = lane%16
  float bv[8];
#pragma unroll
  for (int rr = 0; rr < 8; ++rr) bv[rr] = sBias[8 * half + rr];
  const int obase = (b * Cout + coutBase + 8 * half) * HWo + r0;
  const bool fullTile = (coutBase + 16) <= Cout;   // block-uniform
  if (pixOK) {
    if (fullTile) {
#pragma unroll
      for (int rr = 0; rr < 8; ++rr)
        y[obase + rr * HWo] = act_f<ACT>(acc[rr] + bv[rr]);
    } else {
#pragma unroll
      for (int rr = 0; rr < 8; ++rr)
        if (coutBase + rr + 8 * half < Cout)
          y[obase + rr * HWo] = act_f<ACT>(acc[rr] + bv[rr]);
    }
  }
}

// ================= generic conv (small Cin, e.g. Cin=4) =====================
// K-order: K = ci*ksz*ksz + (kh*ksz+kw) (matches OIHW flat directly).
template<int ACT>
__global__ __launch_bounds__(256)
void convg_k(const float* __restrict__ x, const float* __restrict__ w,
             const float* __restrict__ bias, float* __restrict__ y,
             int B, int Cin, int Hin, int Win,
             int Cout, int Hout, int Wout, int ksz, int stride, int pad)
{
  __shared__ __align__(16) _Float16 sA[2][16 * 32];
  __shared__ float sBias[16];
  const int lane = threadIdx.x & 31;
  const int wave = threadIdx.x >> 5;
  const int half = lane >> 4;
  const int l16  = lane & 15;
  const int coutBase = (int)blockIdx.y * 16;
  const int HWo = Hout * Wout;
  const int HWi = Hin * Win;
  const int Npix = B * HWo;
  const int pix  = ((int)blockIdx.x * 8 + wave) * 16 + l16;
  const bool pixOK = pix < Npix;
  const int pp = pixOK ? pix : 0;
  const int b  = pp / HWo;
  const int r0 = pp - b * HWo;
  const int oh = r0 / Wout, ow = r0 - (r0 / Wout) * Wout;
  const int ohs = oh * stride - pad;
  const int ows = ow * stride - pad;
  const int kk2 = ksz * ksz;
  const int KT  = Cin * kk2;
  const float* __restrict__ xb = x + b * Cin * HWi;

  if (threadIdx.x < 16) {
    int co = coutBase + (int)threadIdx.x;
    sBias[threadIdx.x] = co < Cout ? bias[co] : 0.0f;
  }

  v8f acc = {};
  int c = 0;
  for (int kc = 0; kc < KT; kc += 32, ++c) {
    const int p = c & 1;
    {
      const int m0 = (int)threadIdx.x >> 5, kq = (int)threadIdx.x & 31;
#pragma unroll
      for (int it = 0; it < 2; ++it) {
        int m = m0 + it * 8;
        int K = kc + kq, co = coutBase + m;
        bool ok = (co < Cout) && (K < KT);
        int cc = co < Cout ? co : 0;
        int Kc = K < KT ? K : 0;
        float v = w[cc * KT + Kc];
        sA[p][m * 32 + kq] = (_Float16)(ok ? v : 0.0f);
      }
    }
    __syncthreads();

    const _Float16* rowp = &sA[p][l16 * 32 + 16 * half];
    v8h a0 = *(const v8h*)(rowp);
    v8h a1 = *(const v8h*)(rowp + 8);
    v16h a;
#pragma unroll
    for (int i = 0; i < 8; ++i) { a[i] = a0[i]; a[i + 8] = a1[i]; }

    float tmp[16];
    bool  inb[16];
#pragma unroll
    for (int j = 0; j < 16; ++j) {
      int K  = kc + 16 * half + j;
      int Kc = K < KT ? K : 0;
      int ci = Kc / kk2, rr = Kc - ci * kk2;
      int kh = rr / ksz, kw = rr - (rr / ksz) * ksz;
      int ih = ohs + kh, iw = ows + kw;
      inb[j] = pixOK && (K < KT) && ih >= 0 && ih < Hin && iw >= 0 && iw < Win;
      int ihc = ih < 0 ? 0 : (ih >= Hin ? Hin - 1 : ih);
      int iwc = iw < 0 ? 0 : (iw >= Win ? Win - 1 : iw);
      tmp[j] = xb[ci * HWi + ihc * Win + iwc];           // unconditional
    }
    v16h bf;
#pragma unroll
    for (int j = 0; j < 16; ++j) bf[j] = (_Float16)(inb[j] ? tmp[j] : 0.0f);

    acc = __builtin_amdgcn_wmma_f32_16x16x32_f16(false, a, false, bf,
                                                 (short)0, acc, false, false);
  }

  float bv[8];
#pragma unroll
  for (int rr = 0; rr < 8; ++rr) bv[rr] = sBias[8 * half + rr];
  const int obase = (b * Cout + coutBase + 8 * half) * HWo + r0;
  const bool fullTile = (coutBase + 16) <= Cout;
  if (pixOK) {
    if (fullTile) {
#pragma unroll
      for (int rr = 0; rr < 8; ++rr)
        y[obase + rr * HWo] = act_f<ACT>(acc[rr] + bv[rr]);
    } else {
#pragma unroll
      for (int rr = 0; rr < 8; ++rr)
        if (coutBase + rr + 8 * half < Cout)
          y[obase + rr * HWo] = act_f<ACT>(acc[rr] + bv[rr]);
    }
  }
}

// ================= ConvTranspose2d(k=4,s=2,p=1), parity-split GEMM =========
// torch weight (Cin, Cout, 4, 4). blockIdx.z = output parity (py,px); each
// class taps a fixed 2x2 kernel subset -> GEMM, K = q*Cin + ci (q uniform
// per 32-chunk since Cin%32==0).
__global__ __launch_bounds__(256)
void convt_k(const float* __restrict__ x, const float* __restrict__ w,
             const float* __restrict__ bias, float* __restrict__ y,
             int B, int Cin, int Hin, int Win, int Cout)
{
  __shared__ __align__(16) _Float16 sA[2][16 * 32];
  __shared__ float sBias[16];
  const int Hout = 2 * Hin, Wout = 2 * Win;
  const int py = (int)(blockIdx.z >> 1), px = (int)(blockIdx.z & 1);
  const int kh0 = (py + 1) & 1, kw0 = (px + 1) & 1;
  const int lane = threadIdx.x & 31;
  const int wave = threadIdx.x >> 5;
  const int half = lane >> 4;
  const int l16  = lane & 15;
  const int coutBase = (int)blockIdx.y * 16;
  const int HWi = Hin * Win;
  const int Npix = B * HWi;                      // pixels per parity class
  const int pix  = ((int)blockIdx.x * 8 + wave) * 16 + l16;
  const bool pixOK = pix < Npix;
  const int pp = pixOK ? pix : 0;
  const int b   = pp / HWi;
  const int r0  = pp - b * HWi;
  const int ihb = r0 / Win, iwb = r0 - (r0 / Win) * Win;
  const int oh = 2 * ihb + py, ow = 2 * iwb + px;
  const float* __restrict__ xb = x + b * Cin * HWi;
  const int nChunks = (Cin * 4) >> 5;

  if (threadIdx.x < 16) {
    int co = coutBase + (int)threadIdx.x;
    sBias[threadIdx.x] = co < Cout ? bias[co] : 0.0f;
  }

  v8f acc = {};
  int q = 0, cib = 0;
  for (int c = 0; c < nChunks; ++c) {
    const int p = c & 1;
    const int kh = kh0 + 2 * (q >> 1);
    const int kw = kw0 + 2 * (q & 1);
    {
      const int m0 = (int)threadIdx.x >> 5, kq = (int)threadIdx.x & 31;
#pragma unroll
      for (int it = 0; it < 2; ++it) {
        int m = m0 + it * 8;
        int co = coutBase + m;
        int cc = co < Cout ? co : 0;
        float v = w[((cib + kq) * Cout + cc) * 16 + kh * 4 + kw];
        sA[p][m * 32 + kq] = (_Float16)(co < Cout ? v : 0.0f);
      }
    }
    __syncthreads();

    const _Float16* rowp = &sA[p][l16 * 32 + 16 * half];
    v8h a0 = *(const v8h*)(rowp);
    v8h a1 = *(const v8h*)(rowp + 8);
    v16h a;
#pragma unroll
    for (int i = 0; i < 8; ++i) { a[i] = a0[i]; a[i + 8] = a1[i]; }

    const int ih = (oh + 1 - kh) >> 1;           // exact by parity construction
    const int iw = (ow + 1 - kw) >> 1;
    const bool inb = pixOK && ih >= 0 && ih < Hin && iw >= 0 && iw < Win;
    const int ihc = inb ? ih : 0, iwc = inb ? iw : 0;
    const float* __restrict__ bp = xb + (cib + 16 * half) * HWi + ihc * Win + iwc;
    float tmp[16];
#pragma unroll
    for (int j = 0; j < 16; ++j) tmp[j] = bp[j * HWi];
    v16h bf;
#pragma unroll
    for (int j = 0; j < 16; ++j) bf[j] = (_Float16)(inb ? tmp[j] : 0.0f);

    acc = __builtin_amdgcn_wmma_f32_16x16x32_f16(false, a, false, bf,
                                                 (short)0, acc, false, false);
    cib += 32; if (cib >= Cin) { cib = 0; ++q; }
  }

  float bv[8];
#pragma unroll
  for (int rr = 0; rr < 8; ++rr) bv[rr] = sBias[8 * half + rr];
  const int HWo = Hout * Wout;
  const int obase = (b * Cout + coutBase + 8 * half) * HWo + oh * Wout + ow;
  const bool fullTile = (coutBase + 16) <= Cout;
  if (pixOK) {
    if (fullTile) {
#pragma unroll
      for (int rr = 0; rr < 8; ++rr)
        y[obase + rr * HWo] = acc[rr] + bv[rr];
    } else {
#pragma unroll
      for (int rr = 0; rr < 8; ++rr)
        if (coutBase + rr + 8 * half < Cout)
          y[obase + rr * HWo] = acc[rr] + bv[rr];
    }
  }
}

// ================= GroupNorm (+optional SiLU) ===============================
template<int ACT>
__global__ __launch_bounds__(256)
void groupnorm_k(const float* __restrict__ x, float* __restrict__ y,
                 const float* __restrict__ gamma, const float* __restrict__ beta,
                 int C, int HW, int G)
{
  const int b = (int)blockIdx.x / G, g = (int)blockIdx.x % G;
  const int Cg = C / G;
  const int n = Cg * HW;
  const float* xp = x + (b * C + g * Cg) * HW;
  float* yp = y + (b * C + g * Cg) * HW;
  __shared__ float s1[256], s2[256];
  float sum = 0.0f, sq = 0.0f;
  for (int i = threadIdx.x; i < n; i += 256) { float v = xp[i]; sum += v; sq += v * v; }
  s1[threadIdx.x] = sum; s2[threadIdx.x] = sq;
  __syncthreads();
  for (int o = 128; o > 0; o >>= 1) {
    if ((int)threadIdx.x < o) { s1[threadIdx.x] += s1[threadIdx.x + o];
                                s2[threadIdx.x] += s2[threadIdx.x + o]; }
    __syncthreads();
  }
  const float mean = s1[0] / (float)n;
  const float var  = s2[0] / (float)n - mean * mean;   // biased, matches jnp.var
  const float rstd = rsqrtf(var + 1e-5f);
  for (int i = threadIdx.x; i < n; i += 256) {
    int c = g * Cg + i / HW;
    float v = (xp[i] - mean) * rstd * gamma[c] + beta[c];
    if (ACT) v = silu_f(v);
    yp[i] = v;
  }
}

// ================= residual add + SiLU ======================================
__global__ __launch_bounds__(256)
void add_silu_k(const float* __restrict__ a, const float* __restrict__ b,
                float* __restrict__ o, int n)
{
  int i = (int)blockIdx.x * 256 + (int)threadIdx.x;
  if (i < n) { float v = a[i] + b[i]; o[i] = silu_f(v); }
}

// ================= reparameterize + KL ======================================
__global__ void zero_k(float* p) { *p = 0.0f; }

__global__ __launch_bounds__(256)
void reparam_kl_k(const float* __restrict__ mu, const float* __restrict__ lv,
                  const float* __restrict__ eps, float* __restrict__ z,
                  float* __restrict__ kl, int n)
{
  __shared__ float s[256];
  int i = (int)blockIdx.x * 256 + (int)threadIdx.x;
  float t = 0.0f;
  if (i < n) {
    float m = mu[i], l = lv[i];
    z[i] = m + eps[i] * __expf(0.5f * l);
    t = 0.5f * (__expf(l) + m * m - 1.0f) - 0.5f * l;
  }
  s[threadIdx.x] = t;
  __syncthreads();
  for (int o = 128; o > 0; o >>= 1) {
    if ((int)threadIdx.x < o) s[threadIdx.x] += s[threadIdx.x + o];
    __syncthreads();
  }
  if (threadIdx.x == 0) atomicAdd(kl, s[0] / (float)n);
}

// ================= host-side launch helpers =================================
static void launch_conv(hipStream_t s, const float* x, const float* w, const float* b,
                        float* y, int B, int Cin, int Hin, int Win, int Cout,
                        int ksz, int stride, int pad, int act)
{
  int Hout = (Hin + 2 * pad - ksz) / stride + 1;
  int Wout = (Win + 2 * pad - ksz) / stride + 1;
  int Npix = B * Hout * Wout;
  dim3 g((unsigned)((Npix + 127) / 128), (unsigned)((Cout + 15) / 16), 1);
  dim3 blk(256);
  if (Cin % 32 == 0) {
    if (ksz == 3) {
      if (act == 0)      convf_k<0,3><<<g, blk, 0, s>>>(x, w, b, y, B, Cin, Hin, Win, Cout, Hout, Wout, stride, pad);
      else if (act == 1) convf_k<1,3><<<g, blk, 0, s>>>(x, w, b, y, B, Cin, Hin, Win, Cout, Hout, Wout, stride, pad);
      else               convf_k<2,3><<<g, blk, 0, s>>>(x, w, b, y, B, Cin, Hin, Win, Cout, Hout, Wout, stride, pad);
    } else {
      convf_k<0,1><<<g, blk, 0, s>>>(x, w, b, y, B, Cin, Hin, Win, Cout, Hout, Wout, stride, pad);
    }
  } else {
    if (act == 2) convg_k<2><<<g, blk, 0, s>>>(x, w, b, y, B, Cin, Hin, Win, Cout, Hout, Wout, ksz, stride, pad);
    else          convg_k<0><<<g, blk, 0, s>>>(x, w, b, y, B, Cin, Hin, Win, Cout, Hout, Wout, ksz, stride, pad);
  }
}

static void launch_convt(hipStream_t s, const float* x, const float* w, const float* b,
                         float* y, int B, int Cin, int Hin, int Win, int Cout)
{
  int Npix = B * Hin * Win;
  dim3 g((unsigned)((Npix + 127) / 128), (unsigned)((Cout + 15) / 16), 4);
  convt_k<<<g, dim3(256), 0, s>>>(x, w, b, y, B, Cin, Hin, Win, Cout);
}

static void launch_gn(hipStream_t s, const float* x, float* y, const float* gamma,
                      const float* beta, int B, int C, int H, int W, int act)
{
  int G = C < 32 ? C : 32;
  dim3 g((unsigned)(B * G));
  if (act) groupnorm_k<1><<<g, 256, 0, s>>>(x, y, gamma, beta, C, H * W, G);
  else     groupnorm_k<0><<<g, 256, 0, s>>>(x, y, gamma, beta, C, H * W, G);
}

// res_block: P[pbase..] = conv1.{w,b}, gn1.{g,b}, conv2.{w,b}, gn2.{g,b}[, sc.{w,b}]
static void res_block(hipStream_t s, const float* const* P, int pbase,
                      const float* xin, float* out, float* t1, float* t2,
                      int B, int Cin, int H, int W, int Cout, int stride, bool shortcut)
{
  int Ho = (H + 2 - 3) / stride + 1;
  int Wo = (W + 2 - 3) / stride + 1;
  launch_conv(s, xin, P[pbase + 0], P[pbase + 1], t1, B, Cin, H, W, Cout, 3, stride, 1, 0);
  launch_gn(s, t1, t1, P[pbase + 2], P[pbase + 3], B, Cout, Ho, Wo, 1);
  launch_conv(s, t1, P[pbase + 4], P[pbase + 5], t2, B, Cout, Ho, Wo, Cout, 3, 1, 1, 0);
  launch_gn(s, t2, t2, P[pbase + 6], P[pbase + 7], B, Cout, Ho, Wo, 0);
  const float* resid;
  if (shortcut) {
    launch_conv(s, xin, P[pbase + 8], P[pbase + 9], out, B, Cin, H, W, Cout, 1, stride, 0, 0);
    resid = out;
  } else {
    resid = xin;
  }
  int n = B * Cout * Ho * Wo;
  add_silu_k<<<(unsigned)((n + 255) / 256), 256, 0, s>>>(t2, resid, out, n);
}

// ---------------------------------------------------------------------------
// Input flattening (setup_inputs insertion order): 0:x 1:eps then params:
// 2,3 enc0.{w,b}  4,5 enc0_gn  6..15 enc1  16..25 enc2  26..35 enc3
// 36,37 mu  38,39 logvar  40,41 proj
// 42..49 dec0_res  50,51 dec0_ct  52,53 dec0_gn
// 54..61 dec1_res  62,63 dec1_ct  64,65 dec1_gn
// 66..73 dec2_res  74,75 dec2_ct  76,77 dec2_gn
// 78..85 dec3_res  86,87 dec3_ct  88,89 dec3_gn
// 90,91 out.{w,b}
// Output: x_recon(4194304) | mu(16384) | logvar(16384) | kl(1)
// ---------------------------------------------------------------------------
extern "C" void kernel_launch(void* const* d_in, const int* in_sizes, int n_in,
                              void* d_out, int out_size, void* d_ws, size_t ws_size,
                              hipStream_t stream)
{
  if (n_in < 92) return;
  const float* P[92];
  for (int i = 0; i < 92; ++i) P[i] = (const float*)d_in[i];
  const float* x   = P[0];
  const float* eps = P[1];

  float* out  = (float*)d_out;
  float* mu_o = out + 4194304;       // (8,4,16,32)
  float* lv_o = mu_o + 16384;
  float* kl_o = lv_o + 16384;

  const size_t MB = 1ull << 20;
  char* ws = (char*)d_ws;
  float* buf0 = (float*)(ws);              // up to 128 MB (final decoder tensor)
  float* buf1 = (float*)(ws + 128 * MB);   // up to 32 MB
  float* buf2 = (float*)(ws + 160 * MB);   // temp, up to 32 MB
  float* buf3 = (float*)(ws + 192 * MB);   // temp, up to 32 MB
  float* zb   = (float*)(ws + 224 * MB);   // latent z, 64 KB
  (void)in_sizes; (void)out_size; (void)ws_size;

  const int B = 8;

  // ---- encoder ----
  launch_conv(stream, x, P[2], P[3], buf0, B, 4, 256, 512, 32, 3, 2, 1, 0);   // (8,32,128,256)
  launch_gn(stream, buf0, buf0, P[4], P[5], B, 32, 128, 256, 1);
  res_block(stream, P, 6,  buf0, buf1, buf2, buf3, B, 32, 128, 256, 64, 2, true);   // (8,64,64,128)
  res_block(stream, P, 16, buf1, buf0, buf2, buf3, B, 64, 64, 128, 128, 2, true);   // (8,128,32,64)
  res_block(stream, P, 26, buf0, buf1, buf2, buf3, B, 128, 32, 64, 256, 2, true);   // (8,256,16,32)

  launch_conv(stream, buf1, P[36], P[37], mu_o, B, 256, 16, 32, 4, 3, 1, 1, 0);
  launch_conv(stream, buf1, P[38], P[39], lv_o, B, 256, 16, 32, 4, 3, 1, 1, 0);

  // ---- reparameterize + KL ----
  zero_k<<<1, 1, 0, stream>>>(kl_o);
  const int nlat = 8 * 4 * 16 * 32;   // 16384
  reparam_kl_k<<<(unsigned)((nlat + 255) / 256), 256, 0, stream>>>(mu_o, lv_o, eps, zb, kl_o, nlat);

  // ---- decoder ----
  launch_conv(stream, zb, P[40], P[41], buf0, B, 4, 16, 32, 256, 3, 1, 1, 0);       // (8,256,16,32)

  res_block(stream, P, 42, buf0, buf1, buf2, buf3, B, 256, 16, 32, 256, 1, false);
  launch_convt(stream, buf1, P[50], P[51], buf0, B, 256, 16, 32, 128);              // (8,128,32,64)
  launch_gn(stream, buf0, buf0, P[52], P[53], B, 128, 32, 64, 1);

  res_block(stream, P, 54, buf0, buf1, buf2, buf3, B, 128, 32, 64, 128, 1, false);
  launch_convt(stream, buf1, P[62], P[63], buf0, B, 128, 32, 64, 64);               // (8,64,64,128)
  launch_gn(stream, buf0, buf0, P[64], P[65], B, 64, 64, 128, 1);

  res_block(stream, P, 66, buf0, buf1, buf2, buf3, B, 64, 64, 128, 64, 1, false);
  launch_convt(stream, buf1, P[74], P[75], buf0, B, 64, 64, 128, 32);               // (8,32,128,256)
  launch_gn(stream, buf0, buf0, P[76], P[77], B, 32, 128, 256, 1);

  res_block(stream, P, 78, buf0, buf1, buf2, buf3, B, 32, 128, 256, 32, 1, false);
  launch_convt(stream, buf1, P[86], P[87], buf0, B, 32, 128, 256, 32);              // (8,32,256,512)
  launch_gn(stream, buf0, buf0, P[88], P[89], B, 32, 256, 512, 1);

  // ---- output conv + tanh -> d_out ----
  launch_conv(stream, buf0, P[90], P[91], out, B, 32, 256, 512, 4, 3, 1, 1, 2);
}
